// Model_1039382086439
// MI455X (gfx1250) — matrix operand order repaired and checked by reading, not compiled, not run on hardware
//
#include <hip/hip_runtime.h>
#include <hip/hip_bf16.h>

// ---------------------------------------------------------------------------
// Problem constants
// ---------------------------------------------------------------------------
#define BATCH   64
#define SEQ     720
#define PRED    336
#define CH      862
#define PLEN    96
#define HID     256
#define MAK     25          // moving-average kernel
#define MAH     12          // (MAK-1)/2

#define KDIM    (2*SEQ)     // 1440 : stacked [seasonal ; trend] along K
#define MPAD    384         // PRED padded so 64-row block tiles divide evenly
#define NPADX   896         // CH padded to 128
#define PAIRS   (BATCH*2)   // (batch, variant) pairs
#define CHUNK   8           // pairs staged at once

typedef __attribute__((ext_vector_type(16))) __bf16 v16bf;
typedef __attribute__((ext_vector_type(8)))  __bf16 v8bf;
typedef __attribute__((ext_vector_type(8)))  float  v8f;
typedef __attribute__((ext_vector_type(4)))  int    v4i;

// gfx1250 async global->LDS copy path (ASYNCcnt-tracked), with fallback
#if defined(__gfx1250__) && __has_builtin(__builtin_amdgcn_global_load_async_to_lds_b128)
#define ASYNC_LDS 1
#else
#define ASYNC_LDS 0
#endif

// float -> bf16 round-to-nearest-even
__device__ __forceinline__ __bf16 f2bf(float f) {
    union { float f; unsigned u; } v; v.f = f;
    unsigned r = v.u + 0x7FFFu + ((v.u >> 16) & 1u);
    unsigned short h = (unsigned short)(r >> 16);
    __bf16 o;
    __builtin_memcpy(&o, &h, 2);
    return o;
}

// ---------------------------------------------------------------------------
// 1) RevIN statistics: per (b,c) mean and 1/sqrt(var_ddof1 + 1e-5)
// ---------------------------------------------------------------------------
__global__ void stats_kernel(const float* __restrict__ x,
                             float* __restrict__ meanb,
                             float* __restrict__ rstdb) {
    int gid = blockIdx.x * blockDim.x + threadIdx.x;
    if (gid >= BATCH * CH) return;
    int b = gid / CH, c = gid % CH;
    const float* xp = x + (size_t)b * SEQ * CH + c;
    float s = 0.f, ss = 0.f;
    for (int t = 0; t < SEQ; ++t) {
        float v = xp[(size_t)t * CH];
        s += v; ss += v * v;
    }
    float mean = s * (1.0f / SEQ);
    float var  = (ss - s * mean) * (1.0f / (SEQ - 1)) + 1e-5f;
    meanb[gid] = mean;
    rstdb[gid] = rsqrtf(var);
}

// ---------------------------------------------------------------------------
// 2) Daily index (kept float until the gather cast, like the reference)
// ---------------------------------------------------------------------------
__global__ void idx_kernel(const float* __restrict__ x_mark,
                           float* __restrict__ sidx) {
    int b = threadIdx.x;
    if (b < BATCH)
        sidx[b] = x_mark[((size_t)b * SEQ + (SEQ - 1)) * 4 + 0] * (float)PLEN;
}

// ---------------------------------------------------------------------------
// 3) Stacked weights -> bf16, fragment-major: Wbf[m][k], k<720 => sw else tw
// ---------------------------------------------------------------------------
__global__ void wprep_kernel(const float* __restrict__ sw,
                             const float* __restrict__ tw,
                             __bf16* __restrict__ Wbf) {
    int gid = blockIdx.x * blockDim.x + threadIdx.x;
    if (gid >= MPAD * KDIM) return;
    int m = gid / KDIM, k = gid % KDIM;
    float v = 0.f;
    if (m < PRED)
        v = (k < SEQ) ? sw[(size_t)k * PRED + m] : tw[(size_t)(k - SEQ) * PRED + m];
    Wbf[gid] = f2bf(v);
}

// ---------------------------------------------------------------------------
// 4) Stage B-matrix fragment-major: Xstage[slot][c][k]  (bf16)
//    k in [0,720)    = seasonal(g) = g - movavg(g)
//    k in [720,1440) = movavg(g)
//    g = xn (v=0) or xn - pattern_gather (v=1); rolling 25-tap window.
// ---------------------------------------------------------------------------
__global__ void stage_kernel(const float* __restrict__ x,
                             const float* __restrict__ pattern,
                             const float* __restrict__ sidx,
                             const float* __restrict__ meanb,
                             const float* __restrict__ rstdb,
                             __bf16* __restrict__ Xstage,
                             int pairBase) {
    int c = blockIdx.x * blockDim.x + threadIdx.x;
    if (c >= NPADX) return;
    int slot = blockIdx.y;
    int pair = pairBase + slot;
    int b = pair >> 1, v = pair & 1;
    __bf16* dst = Xstage + (size_t)slot * NPADX * KDIM + (size_t)c * KDIM;

    if (c >= CH) {                       // zero-pad rows
        __bf16 z = f2bf(0.f);
        for (int t = 0; t < KDIM; ++t) dst[t] = z;
        return;
    }

    float mean = meanb[b * CH + c];
    float rstd = rstdb[b * CH + c];
    float s    = sidx[b];
    const float* xp = x + (size_t)b * SEQ * CH + c;

    auto g = [&](int t) -> float {
        float xv = (xp[(size_t)t * CH] - mean) * rstd;
        if (v) {
            int gi = (int)fmodf(s + (float)t, (float)PLEN);
            gi = gi > (PLEN - 1) ? (PLEN - 1) : gi;
            xv -= pattern[(size_t)gi * CH + c];
        }
        return xv;
    };

    float sum = 12.0f * g(0);            // edge-replicated 25-wide window @ t=0
    for (int j = 0; j <= MAH; ++j) sum += g(j);

    for (int t = 0; t < SEQ; ++t) {
        float tr = sum * (1.0f / MAK);
        float se = g(t) - tr;
        dst[t]       = f2bf(se);
        dst[SEQ + t] = f2bf(tr);
        int tin  = t + MAH + 1; if (tin  > SEQ - 1) tin  = SEQ - 1;
        int tout = t - MAH;     if (tout < 0)       tout = 0;
        sum += g(tin) - g(tout);
    }
}

// ---------------------------------------------------------------------------
// 5) WMMA GEMM:  Out[p,c] = sum_k Wbf[p][k] * Xcat[c][k]  (+ sb[p] + tb[p])
//    Block = 8 waves (2x4); wave tile 32x32 (4 WMMA/K-step); block 64x128.
//    LDS tiles are K-contiguous rows so fragments load as ds_load_b128 pairs.
// ---------------------------------------------------------------------------
__global__ void __launch_bounds__(256)
gemm_kernel(const __bf16* __restrict__ Wbf,    // [MPAD][KDIM]
            const __bf16* __restrict__ Xstage, // [slot][NPADX][KDIM]
            const float* __restrict__ sb,
            const float* __restrict__ tb,
            float* __restrict__ pred,
            float* __restrict__ y1,
            int pairBase) {
    __shared__ __attribute__((aligned(16))) __bf16 Asm[64][40];   // [m][k], 80B stride
    __shared__ __attribute__((aligned(16))) __bf16 Bsm[128][40];  // [n][k]

    const int tid  = threadIdx.x;
    const int lane = tid & 31;
    const int wave = tid >> 5;          // 0..7
    const int wm   = wave >> 2;         // 0..1  (M)
    const int wn   = wave & 3;          // 0..3  (N)
    const int half = lane >> 4;         // 0..1
    const int l16  = lane & 15;

    const int mBase = blockIdx.y * 64;   // within MPAD
    const int nBase = blockIdx.x * 128;  // within NPADX
    const int slot  = blockIdx.z;
    const int pair  = pairBase + slot;
    const int b     = pair >> 1;
    const int v     = pair & 1;

    const __bf16* Xs = Xstage + (size_t)slot * NPADX * KDIM;

    // cooperative fill coordinates: one b128 for A, two for B per thread
    const int fm = tid >> 2;            // 0..63
    const int fk = (tid & 3) << 3;      // 0,8,16,24
    const __bf16* gA  = Wbf + (size_t)(mBase + fm) * KDIM + fk;
    const __bf16* gB0 = Xs  + (size_t)(nBase + fm) * KDIM + fk;
    const __bf16* gB1 = Xs  + (size_t)(nBase + 64 + fm) * KDIM + fk;

    // fragment source rows (K-contiguous)
    const __bf16* aRow0 = &Asm[wm * 32 + l16][0];
    const __bf16* aRow1 = &Asm[wm * 32 + 16 + l16][0];
    const __bf16* bRow0 = &Bsm[wn * 32 + l16][0];
    const __bf16* bRow1 = &Bsm[wn * 32 + 16 + l16][0];

    v8f acc00 = {}, acc01 = {}, acc10 = {}, acc11 = {};

    for (int kk = 0; kk < KDIM; kk += 32) {
#if ASYNC_LDS
        __builtin_amdgcn_global_load_async_to_lds_b128(
            (v4i __attribute__((address_space(1)))*)(gA),
            (v4i __attribute__((address_space(3)))*)(&Asm[fm][fk]), 0, 0);
        __builtin_amdgcn_global_load_async_to_lds_b128(
            (v4i __attribute__((address_space(1)))*)(gB0),
            (v4i __attribute__((address_space(3)))*)(&Bsm[fm][fk]), 0, 0);
        __builtin_amdgcn_global_load_async_to_lds_b128(
            (v4i __attribute__((address_space(1)))*)(gB1),
            (v4i __attribute__((address_space(3)))*)(&Bsm[64 + fm][fk]), 0, 0);
#if __has_builtin(__builtin_amdgcn_s_wait_asynccnt)
        __builtin_amdgcn_s_wait_asynccnt(0);
#else
        asm volatile("s_wait_asynccnt 0x0" ::: "memory");
#endif
#else
        *(v8bf*)&Asm[fm][fk]      = *(const v8bf*)gA;
        *(v8bf*)&Bsm[fm][fk]      = *(const v8bf*)gB0;
        *(v8bf*)&Bsm[64 + fm][fk] = *(const v8bf*)gB1;
#endif
        if (kk + 32 < KDIM) {   // pull next tiles toward L2 (global_prefetch_b8)
            __builtin_prefetch(gA + 32, 0, 1);
            __builtin_prefetch(gB0 + 32, 0, 1);
        }
        __syncthreads();

        // ---- fragments: A and B lanes need K {h*8..h*8+7, 16+h*8..}
        v8bf alo0 = *(const v8bf*)(aRow0 + half * 8);
        v8bf ahi0 = *(const v8bf*)(aRow0 + 16 + half * 8);
        v8bf alo1 = *(const v8bf*)(aRow1 + half * 8);
        v8bf ahi1 = *(const v8bf*)(aRow1 + 16 + half * 8);
        v8bf blo0 = *(const v8bf*)(bRow0 + half * 8);
        v8bf bhi0 = *(const v8bf*)(bRow0 + 16 + half * 8);
        v8bf blo1 = *(const v8bf*)(bRow1 + half * 8);
        v8bf bhi1 = *(const v8bf*)(bRow1 + 16 + half * 8);
        v16bf a0 = __builtin_shufflevector(alo0, ahi0, 0,1,2,3,4,5,6,7,8,9,10,11,12,13,14,15);
        v16bf a1 = __builtin_shufflevector(alo1, ahi1, 0,1,2,3,4,5,6,7,8,9,10,11,12,13,14,15);
        v16bf b0 = __builtin_shufflevector(blo0, bhi0, 0,1,2,3,4,5,6,7,8,9,10,11,12,13,14,15);
        v16bf b1 = __builtin_shufflevector(blo1, bhi1, 0,1,2,3,4,5,6,7,8,9,10,11,12,13,14,15);

        acc00 = __builtin_amdgcn_wmma_f32_16x16x32_bf16(false, a0, false, b0, (short)0, acc00, false, false);
        acc01 = __builtin_amdgcn_wmma_f32_16x16x32_bf16(false, a0, false, b1, (short)0, acc01, false, false);
        acc10 = __builtin_amdgcn_wmma_f32_16x16x32_bf16(false, a1, false, b0, (short)0, acc10, false, false);
        acc11 = __builtin_amdgcn_wmma_f32_16x16x32_bf16(false, a1, false, b1, (short)0, acc11, false, false);

        gA += 32; gB0 += 32; gB1 += 32;
        __syncthreads();
    }

    // ---- store D: lane = col, vgpr r -> row r + 8*half; add both biases ----
    float* out = v ? y1 : pred;
    auto store_tile = [&](const v8f& acc, int mf, int nf) {
        int cg = nBase + wn * 32 + nf * 16 + l16;
        if (cg >= CH) return;
#pragma unroll
        for (int r = 0; r < 8; ++r) {
            int p = mBase + wm * 32 + mf * 16 + r + 8 * half;
            if (p < PRED)
                out[((size_t)b * PRED + p) * CH + cg] = acc[r] + sb[p] + tb[p];
        }
    };
    store_tile(acc00, 0, 0);
    store_tile(acc01, 0, 1);
    store_tile(acc10, 1, 0);
    store_tile(acc11, 1, 1);
}

// ---------------------------------------------------------------------------
// 6) Fused mixing MLP + softmax-mean + denorm.  One thread per (b,c).
// ---------------------------------------------------------------------------
__global__ void __launch_bounds__(256)
fuse_kernel(const float* __restrict__ pred,
            const float* __restrict__ y1,
            const float* __restrict__ pattern,
            const float* __restrict__ sidx,
            const float* __restrict__ meanb,
            const float* __restrict__ rstdb,
            const float* __restrict__ w1,
            const float* __restrict__ b1,
            const float* __restrict__ w2,
            const float* __restrict__ b2,
            float* __restrict__ out) {
    __shared__ float sW1[3 * HID];
    __shared__ float sB1[HID];
    __shared__ float sW2[HID * 3];
    __shared__ float sB2[3];
    for (int i = threadIdx.x; i < 3 * HID; i += blockDim.x) sW1[i] = w1[i];
    for (int i = threadIdx.x; i < HID;     i += blockDim.x) sB1[i] = b1[i];
    for (int i = threadIdx.x; i < HID * 3; i += blockDim.x) sW2[i] = w2[i];
    if (threadIdx.x < 3) sB2[threadIdx.x] = b2[threadIdx.x];
    __syncthreads();

    int gid = blockIdx.x * blockDim.x + threadIdx.x;
    if (gid >= BATCH * CH) return;
    int b = gid / CH, c = gid % CH;

    float s    = sidx[b];
    float mean = meanb[gid];
    float sdev = 1.0f / rstdb[gid];     // == sqrt(var + 1e-5)

    float wa0 = 0.f, wa1 = 0.f, wa2 = 0.f;
    for (int p = 0; p < PRED; ++p) {
        size_t o = ((size_t)b * PRED + p) * CH + c;
        float pr = pred[o];
        float yv = y1[o];
        int gi = (int)fmodf(s + (float)SEQ + (float)p, (float)PLEN);
        gi = gi > (PLEN - 1) ? (PLEN - 1) : gi;
        float pt = pattern[(size_t)gi * CH + c];

        float l0 = 0.f, l1 = 0.f, l2 = 0.f;
        for (int j = 0; j < HID; ++j) {
            float h = fmaf(pr, sW1[j],
                      fmaf(yv, sW1[HID + j],
                      fmaf(pt, sW1[2 * HID + j], sB1[j])));
            h = fmaxf(h, 0.f);
            l0 = fmaf(h, sW2[j * 3 + 0], l0);
            l1 = fmaf(h, sW2[j * 3 + 1], l1);
            l2 = fmaf(h, sW2[j * 3 + 2], l2);
        }
        l0 += sB2[0]; l1 += sB2[1]; l2 += sB2[2];
        float mx = fmaxf(l0, fmaxf(l1, l2));
        float e0 = expf(l0 - mx), e1 = expf(l1 - mx), e2 = expf(l2 - mx);
        float inv = 1.0f / (e0 + e1 + e2);
        wa0 += e0 * inv; wa1 += e1 * inv; wa2 += e2 * inv;
    }
    wa0 *= (1.0f / PRED); wa1 *= (1.0f / PRED); wa2 *= (1.0f / PRED);

    for (int p = 0; p < PRED; ++p) {
        size_t o = ((size_t)b * PRED + p) * CH + c;
        float pr = pred[o];
        float yv = y1[o];
        int gi = (int)fmodf(s + (float)SEQ + (float)p, (float)PLEN);
        gi = gi > (PLEN - 1) ? (PLEN - 1) : gi;
        float pt = pattern[(size_t)gi * CH + c];
        float y  = wa0 * pr + wa1 * yv + wa2 * pt;
        out[o] = y * sdev + mean;
    }
}

// ---------------------------------------------------------------------------
// Host launcher
// ---------------------------------------------------------------------------
extern "C" void kernel_launch(void* const* d_in, const int* in_sizes, int n_in,
                              void* d_out, int out_size, void* d_ws, size_t ws_size,
                              hipStream_t stream) {
    const float* x        = (const float*)d_in[0];
    const float* x_mark   = (const float*)d_in[1];
    const float* pattern  = (const float*)d_in[2];
    const float* sw       = (const float*)d_in[3];
    const float* sb       = (const float*)d_in[4];
    const float* tw       = (const float*)d_in[5];
    const float* tb       = (const float*)d_in[6];
    const float* w1       = (const float*)d_in[7];
    const float* b1       = (const float*)d_in[8];
    const float* w2       = (const float*)d_in[9];
    const float* b2       = (const float*)d_in[10];
    float* out            = (float*)d_out;

    // workspace carve-up (256B aligned)
    char* ws = (char*)d_ws;
    auto alloc = [&](size_t bytes) {
        char* p = ws;
        ws += (bytes + 255) & ~(size_t)255;
        return p;
    };
    float*  meanb  = (float*)alloc((size_t)BATCH * CH * 4);
    float*  rstdb  = (float*)alloc((size_t)BATCH * CH * 4);
    float*  sidx   = (float*)alloc(BATCH * 4);
    float*  pred   = (float*)alloc((size_t)BATCH * PRED * CH * 4);
    float*  y1     = (float*)alloc((size_t)BATCH * PRED * CH * 4);
    __bf16* Wbf    = (__bf16*)alloc((size_t)MPAD * KDIM * 2);
    __bf16* Xstage = (__bf16*)alloc((size_t)CHUNK * NPADX * KDIM * 2);
    (void)ws_size; (void)in_sizes; (void)n_in; (void)out_size;

    const int nBC = BATCH * CH;
    stats_kernel<<<(nBC + 255) / 256, 256, 0, stream>>>(x, meanb, rstdb);
    idx_kernel<<<1, 64, 0, stream>>>(x_mark, sidx);
    wprep_kernel<<<(MPAD * KDIM + 255) / 256, 256, 0, stream>>>(sw, tw, Wbf);

    for (int pb = 0; pb < PAIRS; pb += CHUNK) {
        stage_kernel<<<dim3((NPADX + 255) / 256, CHUNK), 256, 0, stream>>>(
            x, pattern, sidx, meanb, rstdb, Xstage, pb);
        gemm_kernel<<<dim3(NPADX / 128, MPAD / 64, CHUNK), 256, 0, stream>>>(
            Wbf, Xstage, sb, tb, pred, y1, pb);
    }

    fuse_kernel<<<(nBC + 255) / 256, 256, 0, stream>>>(
        pred, y1, pattern, sidx, meanb, rstdb, w1, b1, w2, b2, out);
}
